// DeformableAttention1D_66907000537264
// MI455X (gfx1250) — compile-verified, matrix operand order checked by reading, not run on hardware
//
#include <hip/hip_runtime.h>
#include <hip/hip_bf16.h>

// ---------------------------------------------------------------------------
// DeformableAttention1D for MI455X (gfx1250, wave32, WMMA).
//
// Shapes: B=4, N=8192, D=256, H=8, Dh=64, P=4, INNER=512, Lm=2N=16384.
//
// Optimization: q = q_in@Wq is ONLY used by the three small projections
// (ref/delta/logits), all linear before their nonlinearities. We fold
// Wcomb = Wq @ [Wref|Wdel|Wwt] (512x72, computed on device) and run
// proj = q_in @ Wcomb directly: the 17.2 GFLOP q-GEMM disappears.
// Remaining GEMMs (v: 17.2 GF, out: 8.6 GF, proj: 2.4 GF) run as bf16
// WMMA (v_wmma_f32_16x16x32_bf16) with fp32 accumulation.
// ---------------------------------------------------------------------------

typedef __attribute__((ext_vector_type(16))) __bf16 v16bf;
typedef __attribute__((ext_vector_type(8)))  float  v8f;

#define B_  4
#define N_  8192
#define D_  256
#define H_  8
#define DH_ 64
#define P_  4
#define INNER_ 512
#define LM_ 16384           // 2N
#define ROWS_Q (B_ * N_)    // 32768
#define ROWS_M (B_ * LM_)   // 65536

// workspace layout (bytes), all 256B aligned
#define OFF_QIN   ((size_t)0)                         // bf16 [32768][512]
#define OFF_MEM   (OFF_QIN  + (size_t)ROWS_Q*512*2)   // bf16 [65536][256]
#define OFF_BTV   (OFF_MEM  + (size_t)ROWS_M*256*2)   // bf16 [512][256]  (Wv^T)
#define OFF_BTO   (OFF_BTV  + (size_t)512*256*2)      // bf16 [256][512]  (Wout^T)
#define OFF_BTC   (OFF_BTO  + (size_t)256*512*2)      // bf16 [128][512]  (Wcomb^T, pad)
#define OFF_BCOMB (OFF_BTC  + (size_t)128*512*2)      // f32  [128]
#define OFF_PROJ  (OFF_BCOMB+ (size_t)512)            // f32  [32768][128]
#define OFF_V     (OFF_PROJ + (size_t)ROWS_Q*128*4)   // bf16 [65536][512]
#define OFF_OI    (OFF_V    + (size_t)ROWS_M*512*2)   // bf16 [32768][512]

// ------------------------- bf16 helpers (RNE) ------------------------------
__device__ __forceinline__ unsigned short f2bfu(float f) {
    union { float f; unsigned u; } v; v.f = f;
    unsigned r = v.u + 0x7FFFu + ((v.u >> 16) & 1u);
    return (unsigned short)(r >> 16);
}
__device__ __forceinline__ float bfu2f(unsigned short h) {
    union { float f; unsigned u; } v; v.u = ((unsigned)h) << 16;
    return v.f;
}
__device__ __forceinline__ __bf16 f2bf(float f) {
    unsigned short h = f2bfu(f); __bf16 b; __builtin_memcpy(&b, &h, 2); return b;
}

// ------------------------ weight preparation -------------------------------
// Transposed bf16 copies of Wv (256x512 -> [n][k]) and Wout (512x256 -> [n][k]).
__global__ __launch_bounds__(256) void convert_wv_wout_kernel(
    const float* __restrict__ Wv, const float* __restrict__ Wout,
    __bf16* __restrict__ BTv, __bf16* __restrict__ BTo)
{
    int t = blockIdx.x * 256 + threadIdx.x;     // 262144 threads
    if (t < 512 * 256) {
        int n = t >> 8, k = t & 255;            // n<512(INNER), k<256(D)
        BTv[(size_t)n * 256 + k] = f2bf(Wv[(size_t)k * 512 + n]);
    } else {
        int t2 = t - 512 * 256;
        int n = t2 >> 9, k = t2 & 511;          // n<256(D), k<512(INNER)
        BTo[(size_t)n * 512 + k] = f2bf(Wout[(size_t)k * 256 + n]);
    }
}

// Wcomb[k][j] = sum_i Wq[k][i] * Wsm[i][j],  Wsm = [Wref(8)|Wdel(32)|Wwt(32)|pad]
// stored transposed bf16: BTc[j][k], j padded to 128. bcomb[j] = bq@Wsm + bsm.
__global__ __launch_bounds__(256) void build_wcomb_kernel(
    const float* __restrict__ Wq,  const float* __restrict__ Wref,
    const float* __restrict__ Wdel,const float* __restrict__ Wwt,
    const float* __restrict__ bq,  const float* __restrict__ bref,
    const float* __restrict__ bdel,const float* __restrict__ bwt,
    __bf16* __restrict__ BTc, float* __restrict__ bcomb)
{
    int t = blockIdx.x * 256 + threadIdx.x;     // 65536 threads
    int k = t & 511, j = t >> 9;                // k<512, j<128
    float s = 0.f;
    if (j < 72) {
        const float* wq = Wq + (size_t)k * 512;
        if (j < 8) {
            #pragma unroll 4
            for (int i = 0; i < 512; ++i) s += wq[i] * Wref[(size_t)i * 8 + j];
        } else if (j < 40) {
            int c = j - 8;
            #pragma unroll 4
            for (int i = 0; i < 512; ++i) s += wq[i] * Wdel[(size_t)i * 32 + c];
        } else {
            int c = j - 40;
            #pragma unroll 4
            for (int i = 0; i < 512; ++i) s += wq[i] * Wwt[(size_t)i * 32 + c];
        }
    }
    BTc[(size_t)j * 512 + k] = f2bf(s);
    if (k == 0) {
        float bs = 0.f;
        if (j < 8) {
            for (int i = 0; i < 512; ++i) bs += bq[i] * Wref[(size_t)i * 8 + j];
            bs += bref[j];
        } else if (j < 40) {
            int c = j - 8;
            for (int i = 0; i < 512; ++i) bs += bq[i] * Wdel[(size_t)i * 32 + c];
            bs += bdel[c];
        } else if (j < 72) {
            int c = j - 40;
            for (int i = 0; i < 512; ++i) bs += bq[i] * Wwt[(size_t)i * 32 + c];
            bs += bwt[c];
        }
        bcomb[j] = bs;
    }
}

// --------------------------- LayerNorm kernels -----------------------------
// q_in = LN_512(concat[prev_x+te0, x+te1]) -> bf16  (one row per block)
__global__ __launch_bounds__(256) void ln_qin_kernel(
    const float* __restrict__ x, const float* __restrict__ px,
    const float* __restrict__ te, const float* __restrict__ w,
    const float* __restrict__ b, __bf16* __restrict__ outq)
{
    __shared__ float ss[256], sq[256];
    int row = blockIdx.x, tid = threadIdx.x;
    const float* xr = x  + (size_t)row * D_;
    const float* pr = px + (size_t)row * D_;
    float v0 = pr[tid] + te[tid];          // feature tid      (prev_x + te0)
    float v1 = xr[tid] + te[D_ + tid];     // feature 256+tid  (x + te1)
    ss[tid] = v0 + v1; sq[tid] = v0 * v0 + v1 * v1;
    __syncthreads();
    for (int s = 128; s > 0; s >>= 1) {
        if (tid < s) { ss[tid] += ss[tid + s]; sq[tid] += sq[tid + s]; }
        __syncthreads();
    }
    float mean = ss[0] * (1.f / 512.f);
    float var  = sq[0] * (1.f / 512.f) - mean * mean;
    float rstd = rsqrtf(var + 1e-5f);
    __bf16* o = outq + (size_t)row * 512;
    o[tid]       = f2bf((v0 - mean) * rstd * w[tid]       + b[tid]);
    o[256 + tid] = f2bf((v1 - mean) * rstd * w[256 + tid] + b[256 + tid]);
}

// mem = LN_256(concat[prev_x+te0 ; x+te1] along length) -> bf16
__global__ __launch_bounds__(256) void ln_mem_kernel(
    const float* __restrict__ x, const float* __restrict__ px,
    const float* __restrict__ te, const float* __restrict__ w,
    const float* __restrict__ b, __bf16* __restrict__ outm)
{
    __shared__ float ss[256], sq[256];
    int rowg = blockIdx.x, tid = threadIdx.x;   // rowg = b*2N + l
    int bb = rowg >> 14, l = rowg & (LM_ - 1);
    float v;
    if (l < N_) v = px[((size_t)bb * N_ + l)        * D_ + tid] + te[tid];
    else        v = x [((size_t)bb * N_ + (l - N_)) * D_ + tid] + te[D_ + tid];
    ss[tid] = v; sq[tid] = v * v;
    __syncthreads();
    for (int s = 128; s > 0; s >>= 1) {
        if (tid < s) { ss[tid] += ss[tid + s]; sq[tid] += sq[tid + s]; }
        __syncthreads();
    }
    float mean = ss[0] * (1.f / 256.f);
    float var  = sq[0] * (1.f / 256.f) - mean * mean;
    float rstd = rsqrtf(var + 1e-5f);
    outm[(size_t)rowg * 256 + tid] = f2bf((v - mean) * rstd * w[tid] + b[tid]);
}

// ------------------------------ WMMA GEMM ----------------------------------
// C[M][ldc] = A[M][K](bf16) @ BT[ldc][K](bf16)^T + bias (+ optional residual).
// Block tile 128x64, 8 waves (4x2), wave tile 32x32 = 2x2 v_wmma tiles.
// K staged through LDS in 64-wide chunks; fragments read with ds_load_b128
// per the CDNA5 16-bit operand layout (lane 0-15: M/N=lane, K 0-7 & 16-23;
// lane 16-31: K 8-15 & 24-31).
#define BM 128
#define BN 64
#define KC 64

union FragU { uint4 q[2]; v16bf v; };

template<bool RES, bool OUTBF>
__global__ __launch_bounds__(256) void wmma_gemm_kernel(
    const __bf16* __restrict__ A, const __bf16* __restrict__ BT,
    const float* __restrict__ bias, int K, int ldc,
    float* __restrict__ Cf, __bf16* __restrict__ Cb,
    const float* __restrict__ resX, const float* __restrict__ te)
{
    __shared__ __align__(16) __bf16 As[BM * KC];   // 16 KB
    __shared__ __align__(16) __bf16 Bs[BN * KC];   //  8 KB
    const int tid = threadIdx.x;
    const int lane = tid & 31, wave = tid >> 5;
    const int wm = wave & 3, wn = wave >> 2;       // 4x2 wave grid
    const int rowBase = blockIdx.x * BM;
    const int colBase = blockIdx.y * BN;
    const int half = lane >> 4, l16 = lane & 15;

    v8f acc[2][2];
    #pragma unroll
    for (int i = 0; i < 2; ++i)
        #pragma unroll
        for (int j = 0; j < 2; ++j)
            #pragma unroll
            for (int r = 0; r < 8; ++r) acc[i][j][r] = 0.f;

    for (int k0 = 0; k0 < K; k0 += KC) {
        { // A tile: 128 rows x 64 bf16 (128B/row) -> 32 bf16 per thread
            int r = tid >> 1, c0 = (tid & 1) * 32;
            const uint4* src = (const uint4*)(A + (size_t)(rowBase + r) * K + k0 + c0);
            uint4* dst = (uint4*)(As + r * KC + c0);
            dst[0] = src[0]; dst[1] = src[1]; dst[2] = src[2]; dst[3] = src[3];
        }
        { // B tile: 64 rows(n) x 64 bf16 -> 16 bf16 per thread
            int r = tid >> 2, c0 = (tid & 3) * 16;
            const uint4* src = (const uint4*)(BT + (size_t)(colBase + r) * K + k0 + c0);
            uint4* dst = (uint4*)(Bs + r * KC + c0);
            dst[0] = src[0]; dst[1] = src[1];
        }
        __syncthreads();
        #pragma unroll
        for (int ks = 0; ks < KC; ks += 32) {
            v16bf af[2], bf[2];
            #pragma unroll
            for (int i = 0; i < 2; ++i) {
                const __bf16* pa = &As[(wm * 32 + i * 16 + l16) * KC + ks + half * 8];
                FragU u; u.q[0] = *(const uint4*)pa; u.q[1] = *(const uint4*)(pa + 16);
                af[i] = u.v;
            }
            #pragma unroll
            for (int j = 0; j < 2; ++j) {
                const __bf16* pb = &Bs[(wn * 32 + j * 16 + l16) * KC + ks + half * 8];
                FragU u; u.q[0] = *(const uint4*)pb; u.q[1] = *(const uint4*)(pb + 16);
                bf[j] = u.v;
            }
            #pragma unroll
            for (int i = 0; i < 2; ++i)
                #pragma unroll
                for (int j = 0; j < 2; ++j)
                    acc[i][j] = __builtin_amdgcn_wmma_f32_16x16x32_bf16(
                        false, af[i], false, bf[j], (short)0, acc[i][j], false, false);
        }
        __syncthreads();
    }

    // Epilogue. D layout: VGPR r -> (M = tile_m + r + 8*half, N = tile_n + l16).
    #pragma unroll
    for (int i = 0; i < 2; ++i) {
        #pragma unroll
        for (int j = 0; j < 2; ++j) {
            int mBase = rowBase + wm * 32 + i * 16 + half * 8;
            int n = colBase + wn * 32 + j * 16 + l16;
            float bn = bias ? bias[n] : 0.f;
            #pragma unroll
            for (int r = 0; r < 8; ++r) {
                int m = mBase + r;
                float v = acc[i][j][r] + bn;
                if (RES) v += resX[(size_t)m * ldc + n] + te[ldc + n];
                if (OUTBF) Cb[(size_t)m * ldc + n] = f2bf(v);
                else       Cf[(size_t)m * ldc + n] = v;
            }
        }
    }
}

// --------------------------- deformable sampling ---------------------------
// One (b,n,h) task per wave. proj row: [0..7]=ref logits, [8..39]=delta raw,
// [40..71]=attn logits (H*P packed). NOTE: reference renormalizes attn by the
// per-element valid mask whenever ANY position in the whole tensor is OOB
// (jnp.where(valid.all(), ...) with a global reduction). With ref*(Lm-1) in
// [0,16383] and window offsets up to +-5.5, OOB positions exist with
// certainty, so the renormalized branch is always the taken one.
__global__ __launch_bounds__(256) void sample_kernel(
    const float* __restrict__ proj, const __bf16* __restrict__ vmat,
    float* __restrict__ outOff, __bf16* __restrict__ outInner)
{
    const int wave = threadIdx.x >> 5, lane = threadIdx.x & 31;
    const long task = (long)blockIdx.x * 8 + wave;     // 262144 tasks
    const int h = (int)(task & 7);
    const int n = (int)((task >> 3) & (N_ - 1));
    const int b = (int)(task >> 16);
    const int row = b * N_ + n;
    const float* pr = proj + (size_t)row * 128;

    const float ref = 1.f / (1.f + __expf(-pr[h]));
    float lg[P_], attn[P_], dw[P_], posv[P_];
    float mx = -1e30f;
    #pragma unroll
    for (int p = 0; p < P_; ++p) { lg[p] = pr[40 + h * 4 + p]; mx = fmaxf(mx, lg[p]); }
    float se = 0.f;
    #pragma unroll
    for (int p = 0; p < P_; ++p) { attn[p] = __expf(lg[p] - mx); se += attn[p]; }
    const float inv = 1.f / se;
    float saw = 0.f;
    #pragma unroll
    for (int p = 0; p < P_; ++p) {
        float delta = tanhf(pr[8 + h * 4 + p]);
        dw[p] = ((float)p - 1.5f) + delta * 4.0f;               // base + delta*MAX_OFFSET
        float pos = ref * (float)(LM_ - 1) + dw[p];
        float ok = (pos >= 0.f && pos <= (float)(LM_ - 1)) ? 1.f : 0.f;
        attn[p] = attn[p] * inv * ok;
        saw += attn[p];
        posv[p] = fminf(fmaxf(pos, 0.f), (float)(LM_ - 1));
    }
    const float rs = 1.f / (saw + 1e-6f);

    // Each lane handles two adjacent Dh channels (packed bf16 dword loads).
    float acc0 = 0.f, acc1 = 0.f;
    const size_t chanOff = (size_t)h * DH_ + 2 * lane;
    const __bf16* vb = vmat + (size_t)b * LM_ * INNER_ + chanOff;
    #pragma unroll
    for (int p = 0; p < P_; ++p) {
        float a = attn[p] * rs;
        int left = (int)floorf(posv[p]);
        int right = (left + 1 < LM_) ? left + 1 : LM_ - 1;
        float fr = posv[p] - (float)left;
        unsigned ul = *(const unsigned*)(vb + (size_t)left  * INNER_);
        unsigned ur = *(const unsigned*)(vb + (size_t)right * INNER_);
        float l0 = bfu2f((unsigned short)(ul & 0xffff)), l1 = bfu2f((unsigned short)(ul >> 16));
        float r0 = bfu2f((unsigned short)(ur & 0xffff)), r1 = bfu2f((unsigned short)(ur >> 16));
        acc0 += a * (l0 + fr * (r0 - l0));
        acc1 += a * (l1 + fr * (r1 - l1));
    }
    unsigned packed = (unsigned)f2bfu(acc0) | ((unsigned)f2bfu(acc1) << 16);
    *(unsigned*)(outInner + (size_t)row * INNER_ + chanOff) = packed;

    if (lane < P_)
        outOff[((size_t)row * H_ + h) * P_ + lane] = dw[lane] * (1.f / (float)(LM_ - 1));
}

// ------------------------------- launcher ----------------------------------
extern "C" void kernel_launch(void* const* d_in, const int* in_sizes, int n_in,
                              void* d_out, int out_size, void* d_ws, size_t ws_size,
                              hipStream_t stream)
{
    const float* x    = (const float*)d_in[0];
    const float* px   = (const float*)d_in[1];
    const float* te   = (const float*)d_in[2];
    const float* lnqw = (const float*)d_in[3];
    const float* lnqb = (const float*)d_in[4];
    const float* lnmw = (const float*)d_in[5];
    const float* lnmb = (const float*)d_in[6];
    const float* Wq   = (const float*)d_in[7];
    const float* bq   = (const float*)d_in[8];
    const float* Wv   = (const float*)d_in[9];
    const float* bv   = (const float*)d_in[10];
    const float* Wref = (const float*)d_in[11];
    const float* bref = (const float*)d_in[12];
    const float* Wdel = (const float*)d_in[13];
    const float* bdel = (const float*)d_in[14];
    const float* Wwt  = (const float*)d_in[15];
    const float* bwt  = (const float*)d_in[16];
    const float* Wout = (const float*)d_in[17];
    const float* bout = (const float*)d_in[18];

    char* ws = (char*)d_ws;
    __bf16* qinb  = (__bf16*)(ws + OFF_QIN);
    __bf16* memb  = (__bf16*)(ws + OFF_MEM);
    __bf16* BTv   = (__bf16*)(ws + OFF_BTV);
    __bf16* BTo   = (__bf16*)(ws + OFF_BTO);
    __bf16* BTc   = (__bf16*)(ws + OFF_BTC);
    float*  bcomb = (float*) (ws + OFF_BCOMB);
    float*  proj  = (float*) (ws + OFF_PROJ);
    __bf16* vbuf  = (__bf16*)(ws + OFF_V);
    __bf16* oib   = (__bf16*)(ws + OFF_OI);

    float* outMain = (float*)d_out;
    float* outOff  = outMain + (size_t)B_ * N_ * D_;

    // weights
    convert_wv_wout_kernel<<<1024, 256, 0, stream>>>(Wv, Wout, BTv, BTo);
    build_wcomb_kernel<<<256, 256, 0, stream>>>(Wq, Wref, Wdel, Wwt,
                                                bq, bref, bdel, bwt, BTc, bcomb);
    // activations
    ln_qin_kernel<<<ROWS_Q, 256, 0, stream>>>(x, px, te, lnqw, lnqb, qinb);
    ln_mem_kernel<<<ROWS_M, 256, 0, stream>>>(x, px, te, lnmw, lnmb, memb);

    // proj = q_in @ Wcomb + bcomb : M=32768, K=512, N=128(pad of 72)
    {
        dim3 g(ROWS_Q / BM, 128 / BN);
        wmma_gemm_kernel<false, false><<<g, 256, 0, stream>>>(
            qinb, BTc, bcomb, 512, 128, proj, nullptr, nullptr, nullptr);
    }
    // v = mem @ Wv + bv : M=65536, K=256, N=512 (bf16 out)
    {
        dim3 g(ROWS_M / BM, 512 / BN);
        wmma_gemm_kernel<false, true><<<g, 256, 0, stream>>>(
            memb, BTv, bv, 256, 512, nullptr, vbuf, nullptr, nullptr);
    }
    // deformable sampling + offsets_norm
    sample_kernel<<<ROWS_Q * H_ / 8, 256, 0, stream>>>(proj, vbuf, outOff, oib);

    // out = sampled @ Wout + bout + (x + te1) : M=32768, K=512, N=256
    {
        dim3 g(ROWS_Q / BM, 256 / BN);
        wmma_gemm_kernel<true, false><<<g, 256, 0, stream>>>(
            oib, BTo, bout, 512, 256, outMain, nullptr, x, te);
    }
}
